// ProxyNCA_prob_mixup_40664750359181
// MI455X (gfx1250) — matrix operand order repaired and checked by reading, not compiled, not run on hardware
//
#include <hip/hip_runtime.h>
#include <hip/hip_bf16.h>

// ProxyNCA loss, fused GEMM + logsumexp for MI455X (gfx1250, wave32, WMMA).
//
// loss = mean_i ( D[i,T[i]] + log sum_j exp(-D[i,j]) )
// D[i,j] = max(18 - 2*s_ij, 0),  s = (3*norm(X)) @ (3*norm(P))^T
// Since ||rows|| == 3 exactly, -D in [-36, 0]: no online-max needed for the
// logsumexp (exp(-36) ~ 2e-16 is far from f32 underflow).
//
// exp is computed as exp2 with folded constants:
//   exp(2s - 18) = exp2(s*C1 + C2), C1 = 2*log2(e), C2 = -18*log2(e)
// and the D>=0 clamp becomes min(s*C1 + C2, 0) (log2(e) > 0).

#define NB_CLASSES 65536
#define SZ_EMBED   64
#define NPTS       4096
#define PSCALE     3.0f

#define ROW_BLOCKS (NPTS / 16)          // 256 row blocks of 16 rows
#define COL_PARTS  16                   // column partitions per row block
#define COLS_PER_PART (NB_CLASSES / COL_PARTS)  // 4096
#define TILES_PER_PART (COLS_PER_PART / 16)     // 256 tiles of 16 cols

#define EXP_C1 2.8853900817779268f      //  2 * log2(e)
#define EXP_C2 (-25.968510735997341f)   // -18 * log2(e)

typedef _Float16 v16h __attribute__((ext_vector_type(16)));
typedef _Float16 v8h  __attribute__((ext_vector_type(8)));
typedef float    v8f  __attribute__((ext_vector_type(8)));

// ---------------------------------------------------------------------------
// Zero the per-row sum-of-exp accumulator (atomicAdd target).
// ---------------------------------------------------------------------------
__global__ void pnca_zero_kernel(float* __restrict__ p, int n)
{
    int i = blockIdx.x * blockDim.x + threadIdx.x;
    if (i < n) p[i] = 0.0f;
}

// ---------------------------------------------------------------------------
// Normalize rows (p=2, eps=1e-12), scale by 3, convert to f16.
// One wave (32 lanes) per 64-element row; rows [0,NB) = proxies, rest = X.
// ---------------------------------------------------------------------------
__global__ void pnca_normalize_f16_kernel(const float* __restrict__ X,
                                          const float* __restrict__ P,
                                          _Float16* __restrict__ X16,
                                          _Float16* __restrict__ P16)
{
    const int lane  = threadIdx.x & 31;
    const int wib   = threadIdx.x >> 5;
    const long row  = (long)blockIdx.x * (blockDim.x >> 5) + wib;
    const long total = (long)NB_CLASSES + NPTS;
    if (row >= total) return;

    const float* src;
    _Float16*    dst;
    if (row < NB_CLASSES) { src = P + row * SZ_EMBED;                 dst = P16 + row * SZ_EMBED; }
    else                  { src = X + (row - NB_CLASSES) * SZ_EMBED;  dst = X16 + (row - NB_CLASSES) * SZ_EMBED; }

    float x0 = src[lane];
    float x1 = src[lane + 32];
    float ss = fmaf(x0, x0, x1 * x1);
#pragma unroll
    for (int m = 1; m < 32; m <<= 1) ss += __shfl_xor(ss, m, 32);
    float s = PSCALE / fmaxf(sqrtf(ss), 1e-12f);
    dst[lane]      = (_Float16)(x0 * s);
    dst[lane + 32] = (_Float16)(x1 * s);
}

// ---------------------------------------------------------------------------
// Target term in full f32: tgt[i] = max(18 - 2 * 9 * <x_i, p_T[i]>_hat, 0).
// ---------------------------------------------------------------------------
__global__ void pnca_target_kernel(const float* __restrict__ X,
                                   const float* __restrict__ P,
                                   const int*   __restrict__ T,
                                   float*       __restrict__ tgt)
{
    int i = blockIdx.x * blockDim.x + threadIdx.x;
    if (i >= NPTS) return;
    const float* x = X + (long)i * SZ_EMBED;
    const float* p = P + (long)T[i] * SZ_EMBED;
    float dxp = 0.f, dxx = 0.f, dpp = 0.f;
#pragma unroll 8
    for (int k = 0; k < SZ_EMBED; ++k) {
        float a = x[k], b = p[k];
        dxp = fmaf(a, b, dxp);
        dxx = fmaf(a, a, dxx);
        dpp = fmaf(b, b, dpp);
    }
    float s = (PSCALE * PSCALE) * dxp /
              (fmaxf(sqrtf(dxx), 1e-12f) * fmaxf(sqrtf(dpp), 1e-12f));
    tgt[i] = fmaxf(fmaf(-2.0f, s, 2.0f * PSCALE * PSCALE), 0.0f);
}

// ---------------------------------------------------------------------------
// Fused WMMA GEMM + sum-of-exp, software-pipelined.
//
// One wave owns a 16-row block and one column partition (4096 cols = 256
// tiles). A (16x64 f16) is loop-invariant in registers (two v16h). Per tile:
// issue B loads for tile t+1, then two v_wmma_f32_16x16x32_f16 (K=64) and
// 8 exp2 per lane for tile t — the exp chain hides the L2 load latency.
// The 8 waves of a block scan identical B tiles (same colPart), so the WGP$
// serves 7/8 of the B reads. One-tile address overrun past P16 lands in the
// adjacent X16 workspace region (valid memory; values unused).
//
// VGPR layouts used here (A/B k order is this code's contiguous one, not the ISA table order):
//   A 16x32: lane l -> M=l&15; K run  kb..kb+15, kb=(l<16?0:16)  (matches B)
//   B 32x16: lane l -> N=l&15; K run  kb..kb+15, kb=(l<16?0:16)  (contiguous!)
//   C/D:     vgpr r, lane l -> (M = r + 8*(l>>4), N = l&15)
// ---------------------------------------------------------------------------
__launch_bounds__(256)
__global__ void pnca_lse_wmma_kernel(const _Float16* __restrict__ X16,
                                     const _Float16* __restrict__ P16,
                                     float* __restrict__ sumexp)
{
    const int lane = threadIdx.x & 31;
    const int wave = blockIdx.x * (blockDim.x >> 5) + (threadIdx.x >> 5);
    const int rowBlock = wave & (ROW_BLOCKS - 1);  // waves in a block share colPart
    const int colPart  = wave >> 8;                // log2(ROW_BLOCKS) == 8
    const int m  = lane & 15;
    const int hi = lane >> 4;

    // ---- A matrix: 16 rows x K=64, split into two 16x32 operands ----
    const _Float16* xrow = X16 + (size_t)(rowBlock * 16 + m) * SZ_EMBED;
    const int ka = hi ? 16 : 0;
    v8h a0lo = *(const v8h*)(xrow + ka);
    v8h a0hi = *(const v8h*)(xrow + ka + 8);
    v8h a1lo = *(const v8h*)(xrow + ka + 32);
    v8h a1hi = *(const v8h*)(xrow + ka + 40);
    v16h a0 = __builtin_shufflevector(a0lo, a0hi, 0,1,2,3,4,5,6,7,8,9,10,11,12,13,14,15);
    v16h a1 = __builtin_shufflevector(a1lo, a1hi, 0,1,2,3,4,5,6,7,8,9,10,11,12,13,14,15);

    const int kb = hi ? 16 : 0;
    // This lane's B pointer for tile 0 of its partition.
    const _Float16* prow =
        P16 + (size_t)(colPart * COLS_PER_PART + m) * SZ_EMBED + kb;
    const size_t tileStep = (size_t)16 * SZ_EMBED;   // 16 proxy rows per tile

    float sums[8] = {0.f, 0.f, 0.f, 0.f, 0.f, 0.f, 0.f, 0.f};

    // Prologue: load tile 0.
    v16h b0 = *(const v16h*)(prow);
    v16h b1 = *(const v16h*)(prow + 32);

    for (int t = 0; t < TILES_PER_PART; ++t) {
        // Issue next tile's loads first so they overlap the exp chain below.
        const _Float16* pnext = prow + tileStep;
        v16h nb0 = *(const v16h*)(pnext);
        v16h nb1 = *(const v16h*)(pnext + 32);
        __builtin_prefetch(pnext + tileStep, 0, 3);  // global_prefetch_b8 (near)
        prow = pnext;

        v8f acc = {};
        acc = __builtin_amdgcn_wmma_f32_16x16x32_f16(false, a0, false, b0,
                                                     (short)0, acc, false, false);
        acc = __builtin_amdgcn_wmma_f32_16x16x32_f16(false, a1, false, b1,
                                                     (short)0, acc, false, false);

#pragma unroll
        for (int r = 0; r < 8; ++r) {
            // exp(min(2s-18,0)) == exp2(min(s*C1 + C2, 0)); v_exp_f32 is base-2
            float nd = fminf(fmaf(EXP_C1, acc[r], EXP_C2), 0.0f);
            sums[r] += __builtin_amdgcn_exp2f(nd);   // v_exp_f32 (TRANS)
        }

        b0 = nb0;
        b1 = nb1;
    }

    // Row totals: reduce each vgpr slot across its 16-lane half, then one
    // atomicAdd per row from the half's lane 0.
#pragma unroll
    for (int r = 0; r < 8; ++r) {
        float v = sums[r];
#pragma unroll
        for (int msk = 1; msk < 16; msk <<= 1) v += __shfl_xor(v, msk, 16);
        if (m == 0) atomicAdd(&sumexp[rowBlock * 16 + r + 8 * hi], v);
    }
}

// ---------------------------------------------------------------------------
// Final: out = mean_i( tgt[i] + log(sumexp[i]) )
// ---------------------------------------------------------------------------
__global__ void pnca_final_kernel(const float* __restrict__ tgt,
                                  const float* __restrict__ sumexp,
                                  float* __restrict__ out)
{
    __shared__ float sh[256];
    float acc = 0.0f;
    for (int i = threadIdx.x; i < NPTS; i += blockDim.x)
        acc += tgt[i] + __logf(sumexp[i]);
    sh[threadIdx.x] = acc;
    __syncthreads();
    for (int s = 128; s > 0; s >>= 1) {
        if (threadIdx.x < (unsigned)s) sh[threadIdx.x] += sh[threadIdx.x + s];
        __syncthreads();
    }
    if (threadIdx.x == 0) out[0] = sh[0] * (1.0f / NPTS);
}

// ---------------------------------------------------------------------------
extern "C" void kernel_launch(void* const* d_in, const int* in_sizes, int n_in,
                              void* d_out, int out_size, void* d_ws, size_t ws_size,
                              hipStream_t stream)
{
    const float* X = (const float*)d_in[0];   // [4096, 64]
    const float* P = (const float*)d_in[1];   // [65536, 64]
    /* d_in[2] = indices (unused by the math) */
    const int*   T = (const int*)d_in[3];     // [4096]
    float* out = (float*)d_out;

    // Workspace layout (~8.6 MB):
    char* ws = (char*)d_ws;
    _Float16* P16    = (_Float16*)ws;                                           // 8 MB
    _Float16* X16    = (_Float16*)(ws + (size_t)NB_CLASSES * SZ_EMBED * 2);     // 512 KB
    float*    sumexp = (float*)(ws + (size_t)(NB_CLASSES + NPTS) * SZ_EMBED * 2);
    float*    tgt    = sumexp + NPTS;

    // 1) zero accumulators
    pnca_zero_kernel<<<(NPTS + 255) / 256, 256, 0, stream>>>(sumexp, NPTS);

    // 2) normalize + f16 convert (one wave per row; 8 waves / 256-thread block)
    {
        int rows = NB_CLASSES + NPTS;
        pnca_normalize_f16_kernel<<<(rows + 7) / 8, 256, 0, stream>>>(X, P, X16, P16);
    }

    // 3) target distances in full f32
    pnca_target_kernel<<<(NPTS + 255) / 256, 256, 0, stream>>>(X, P, T, tgt);

    // 4) fused WMMA + logsumexp accumulation
    //    4096 waves = 256 row blocks x 16 column partitions, 8 waves/block
    pnca_lse_wmma_kernel<<<(ROW_BLOCKS * COL_PARTS) / 8, 256, 0, stream>>>(X16, P16, sumexp);

    // 5) final mean
    pnca_final_kernel<<<1, 256, 0, stream>>>(tgt, sumexp, out);
}
